// Glasso_NeuralODE_67894843015289
// MI455X (gfx1250) — matrix operand order repaired and checked by reading, not compiled, not run on hardware
//
#include <hip/hip_runtime.h>

// ---------------------------------------------------------------------------
// Glasso Neural-ODE RK4 integrator for MI455X (gfx1250, wave32, WMMA).
//
// Strategy (see analysis): batch rows are independent ODEs -> persistent
// kernel, 16 rows per workgroup, 16 workgroups x 512 threads. All 199 RK4
// steps run inside one kernel with only workgroup barriers. GEMMs use
// v_wmma_f32_16x16x32_bf16 with split-precision (hi/lo bf16, 3 products)
// for ~fp32 accuracy. Weights pre-swizzled into WMMA B-tile layout in d_ws
// (L2-resident, fully coalesced b128 loads). Activations staged in LDS
// (XOR-swizzled, bank-conflict-free ds_load_b128 in a contiguous-K A-layout).
// ---------------------------------------------------------------------------

typedef __bf16 bf16_t;
typedef __bf16 v16bf __attribute__((ext_vector_type(16)));
typedef __bf16 v8bf  __attribute__((ext_vector_type(8)));
typedef float  v8f   __attribute__((ext_vector_type(8)));

#define NN      512
#define BATCH   256
#define TSTEPS  200
#define ROWS    16            // batch rows per workgroup (one WMMA M tile)
#define THREADS 512           // 16 waves; each wave owns 32 output columns
#define NELEM   (NN * NN)     // 262144 elements per weight matrix

// d_ws layout (bf16 elements, swizzled tile order):
//   [0*NELEM] Wm_hi  [1*NELEM] Wm_lo  [2*NELEM] W2_hi  [3*NELEM] W2_lo

__device__ inline v8f wmma_bf16(v16bf a, v16bf b, v8f c) {
  // (neg_a, A, neg_b, B, c_mod, C, reuse_a, reuse_b)
  return __builtin_amdgcn_wmma_f32_16x16x32_bf16(false, a, false, b,
                                                 (short)0, c, false, false);
}

__device__ inline v8f splat8(float x) {
  v8f r;
#pragma unroll
  for (int j = 0; j < 8; ++j) r[j] = x;
  return r;
}

// LDS activation staging: row m (0..15), column n (0..511), bf16.
// Chunk (8 elements = 16B) index is XOR-swizzled with the row so that the
// 16 lanes of a WMMA A-load hit 16 distinct bank groups (zero conflicts)
// with zero padding (4 buffers x 16KB = 64KB LDS exactly).
__device__ inline int stage_idx(int m, int n) {
  return (m << 9) + ((((n >> 3) ^ m) & 63) << 3) + (n & 7);
}

__device__ inline void stage_store(bf16_t* H, bf16_t* L, int m, int n, float x) {
  int idx = stage_idx(m, n);
  bf16_t h = (bf16_t)x;
  H[idx] = h;
  L[idx] = (bf16_t)(x - (float)h);
}

// A-matrix 16x32 bf16 load: lane L -> row m = L&15,
// khalf = L>>4; elements hold K in [khalf*16, khalf*16+16) (matches B packing).
// Two contiguous 16B chunks per lane -> two ds_load_b128.
__device__ inline v16bf load_a(const bf16_t* S, int m, int kt, int kh) {
  int c1 = (kt << 2) + (kh << 1); // absolute 8-elem chunk index of first half
  int c2 = c1 + 1;              // +8 elements
  v8bf a0 = *(const v8bf*)(S + (m << 9) + (((c1 ^ m) & 63) << 3));
  v8bf a1 = *(const v8bf*)(S + (m << 9) + (((c2 ^ m) & 63) << 3));
  return __builtin_shufflevector(a0, a1, 0, 1, 2, 3, 4, 5, 6, 7,
                                 8, 9, 10, 11, 12, 13, 14, 15);
}

// B-matrix 32x16 bf16 load from pre-swizzled weights: per (nt,kt) tile the
// init kernel stored each lane's 16 contiguous K values (lane = khalf*16 + n%16,
// same contiguous-K order as load_a: lanes 0-15 K=0..15, lanes 16-31 K=16..31, n = lane&15).
// -> wave reads 1KB fully contiguous: two global_load_b128 per lane.
__device__ inline v16bf load_b(const bf16_t* W, int nt, int kt, int lane) {
  const bf16_t* p = W + ((((nt << 4) + kt) << 5) + lane) * 16;
  v8bf x0 = *(const v8bf*)(p);
  v8bf x1 = *(const v8bf*)(p + 8);
  return __builtin_shufflevector(x0, x1, 0, 1, 2, 3, 4, 5, 6, 7,
                                 8, 9, 10, 11, 12, 13, 14, 15);
}

// One GEMM phase for this wave's two 16x16 output tiles, K = 512.
// Split-precision: D += Ah*Bh + Ah*Bl + Al*Bh (f32 accumulate).
__device__ inline void gemm_pair(const bf16_t* SH, const bf16_t* SL,
                                 const bf16_t* WH, const bf16_t* WL,
                                 int am, int kh, int lane, int nt0, int nt1,
                                 v8f& acc0, v8f& acc1) {
#pragma unroll 4
  for (int kt = 0; kt < 16; ++kt) {
    v16bf ah = load_a(SH, am, kt, kh);
    v16bf al = load_a(SL, am, kt, kh);
    v16bf bh0 = load_b(WH, nt0, kt, lane);
    v16bf bl0 = load_b(WL, nt0, kt, lane);
    acc0 = wmma_bf16(ah, bh0, acc0);
    acc0 = wmma_bf16(ah, bl0, acc0);
    acc0 = wmma_bf16(al, bh0, acc0);
    v16bf bh1 = load_b(WH, nt1, kt, lane);
    v16bf bl1 = load_b(WL, nt1, kt, lane);
    acc1 = wmma_bf16(ah, bh1, acc1);
    acc1 = wmma_bf16(ah, bl1, acc1);
    acc1 = wmma_bf16(al, bh1, acc1);
  }
}

// RK4 post-processing for one output tile column. This thread owns elements
// (m = mbase+v, n) of k (D-matrix layout), of y and of the RK4 accumulator.
__device__ inline void rk4_post(int e, float dt, float* y, float* ac,
                                const v8f& k, int mbase, int n, int b0, int s,
                                float* out, bf16_t* AH, bf16_t* AL) {
#pragma unroll
  for (int v = 0; v < 8; ++v) {
    float kv = k[v];
    float w = (e == 1 || e == 2) ? 2.0f : 1.0f;
    float a = (e == 0) ? kv : ac[v] + w * kv;
    ac[v] = a;
    int m = mbase + v;
    float z;
    if (e == 0)      z = y[v] + 0.5f * dt * kv;   // input for k2
    else if (e == 1) z = y[v] + 0.5f * dt * kv;   // input for k3
    else if (e == 2) z = y[v] + dt * kv;          // input for k4
    else {                                        // finalize step
      z = y[v] + (dt * (1.0f / 6.0f)) * a;
      y[v] = z;
      out[((size_t)(b0 + m) * TSTEPS + (s + 1)) * NN + n] = z;
    }
    stage_store(AH, AL, m, n, z);                 // next GEMM1 input
  }
}

// ---------------------------------------------------------------------------
// Init kernel: Wm = W1*mask and W2 -> split hi/lo bf16, stored in the
// swizzled B-tile layout. Runs once per launch (2MB, negligible).
// ---------------------------------------------------------------------------
__global__ void weights_init_kernel(const float* __restrict__ W1,
                                    const float* __restrict__ mask,
                                    const float* __restrict__ W2,
                                    bf16_t* __restrict__ ws) {
  int i = blockIdx.x * blockDim.x + threadIdx.x;   // i = n*512 + k
  if (i >= NELEM) return;
  int n = i >> 9, k = i & 511;
  int nt = n >> 4, nl = n & 15, kt = k >> 5, kr = k & 31;
  int lane = ((kr >> 4) << 4) + nl;                 // khalf*16 + n%16
  int sidx = ((((nt << 4) + kt) << 5) + lane) * 16 + (kr & 15);
  float wm = W1[i] * mask[i];
  bf16_t h1 = (bf16_t)wm;
  ws[0 * NELEM + sidx] = h1;
  ws[1 * NELEM + sidx] = (bf16_t)(wm - (float)h1);
  float w2 = W2[i];
  bf16_t h2 = (bf16_t)w2;
  ws[2 * NELEM + sidx] = h2;
  ws[3 * NELEM + sidx] = (bf16_t)(w2 - (float)h2);
}

// ---------------------------------------------------------------------------
// Persistent ODE kernel: one workgroup = 16 batch rows, all 200 steps.
// ---------------------------------------------------------------------------
__global__ __launch_bounds__(THREADS, 1)
void ode_rk4_kernel(const float* __restrict__ x_init,
                    const float* __restrict__ ts,
                    const float* __restrict__ b1,
                    const float* __restrict__ b2,
                    const bf16_t* __restrict__ ws,
                    float* __restrict__ out) {
  __shared__ bf16_t AH[ROWS * NN];   // GEMM1 input, hi
  __shared__ bf16_t AL[ROWS * NN];   // GEMM1 input, lo
  __shared__ bf16_t HH[ROWS * NN];   // GEMM2 input (tanh hidden), hi
  __shared__ bf16_t HL[ROWS * NN];   // GEMM2 input, lo

  const int tid  = threadIdx.x;
  const int wave = tid >> 5;
  const int lane = tid & 31;
  const int nl   = lane & 15;        // A-row for WMMA A loads; n%16 for B/C/D
  const int kh   = lane >> 4;        // K-half for A loads; m-half for C/D
  const int b0   = blockIdx.x * ROWS;
  const int nt0  = wave * 2, nt1 = wave * 2 + 1;   // this wave's N tiles
  const int n0   = (nt0 << 4) + nl, n1 = (nt1 << 4) + nl;
  const int mrow = kh << 3;          // D-layout M base for this lane

  const bf16_t* WmH = ws;
  const bf16_t* WmL = ws + 1 * NELEM;
  const bf16_t* W2H = ws + 2 * NELEM;
  const bf16_t* W2L = ws + 3 * NELEM;

  const float b1v0 = b1[n0], b1v1 = b1[n1];
  const float b2v0 = b2[n0], b2v1 = b2[n1];

  // Per-thread ODE state: y[m][n] for m = mrow..mrow+7, n in {n0,n1}.
  float y0[8], y1[8], ac0[8], ac1[8];

#pragma unroll
  for (int v = 0; v < 8; ++v) {
    int m = mrow + v;
    float x0 = x_init[(b0 + m) * NN + n0];
    float x1 = x_init[(b0 + m) * NN + n1];
    y0[v] = x0; y1[v] = x1; ac0[v] = 0.0f; ac1[v] = 0.0f;
    out[((size_t)(b0 + m) * TSTEPS) * NN + n0] = x0;   // traj[:,0,:]
    out[((size_t)(b0 + m) * TSTEPS) * NN + n1] = x1;
    stage_store(AH, AL, m, n0, x0);                    // z for k1 of step 0
    stage_store(AH, AL, m, n1, x1);
  }

  for (int s = 0; s < TSTEPS - 1; ++s) {
    const float dt = ts[s + 1] - ts[s];
    for (int e = 0; e < 4; ++e) {
      __syncthreads();                           // A-stage writes visible
      // GEMM1: hidden = tanh(Z @ Wm^T + b1)
      v8f h0 = splat8(b1v0), h1 = splat8(b1v1);
      gemm_pair(AH, AL, WmH, WmL, nl, kh, lane, nt0, nt1, h0, h1);
#pragma unroll
      for (int v = 0; v < 8; ++v) {
        stage_store(HH, HL, mrow + v, n0, tanhf(h0[v]));
        stage_store(HH, HL, mrow + v, n1, tanhf(h1[v]));
      }
      __syncthreads();                           // H-stage writes visible
      // GEMM2: k = H @ W2^T + b2, then RK4 combine + restage next input
      v8f k0 = splat8(b2v0), k1 = splat8(b2v1);
      gemm_pair(HH, HL, W2H, W2L, nl, kh, lane, nt0, nt1, k0, k1);
      rk4_post(e, dt, y0, ac0, k0, mrow, n0, b0, s, out, AH, AL);
      rk4_post(e, dt, y1, ac1, k1, mrow, n1, b0, s, out, AH, AL);
    }
  }
}

extern "C" void kernel_launch(void* const* d_in, const int* in_sizes, int n_in,
                              void* d_out, int out_size, void* d_ws, size_t ws_size,
                              hipStream_t stream) {
  const float* x_init = (const float*)d_in[0];
  const float* t_span = (const float*)d_in[1];
  const float* W1     = (const float*)d_in[2];
  const float* b1     = (const float*)d_in[3];
  const float* W2     = (const float*)d_in[4];
  const float* b2     = (const float*)d_in[5];
  const float* mask   = (const float*)d_in[6];
  bf16_t* ws  = (bf16_t*)d_ws;     // needs 4*NELEM*2 = 2 MB scratch
  float*  out = (float*)d_out;

  weights_init_kernel<<<NELEM / 256, 256, 0, stream>>>(W1, mask, W2, ws);
  ode_rk4_kernel<<<BATCH / ROWS, THREADS, 0, stream>>>(x_init, t_span,
                                                       b1, b2, ws, out);
}
